// GQAttention_88785563942915
// MI455X (gfx1250) — matrix operand order repaired and checked by reading, not compiled, not run on hardware
//
#include <hip/hip_runtime.h>
#include <hip/hip_bf16.h>
#include <math.h>

// ---------------------------------------------------------------------------
// Problem constants (must match reference)
// ---------------------------------------------------------------------------
#define DIM_     4096
#define NH_      32
#define NKV_     8
#define HD_      128           // head dim
#define KVD_     1024          // NKV_ * HD_
#define QKVD_    6144          // DIM_ + 2*KVD_
#define B_       2
#define T_       1024
#define MTOT_    2048          // B_ * T_

// ---------------------------------------------------------------------------
// WMMA / TDM types (CDNA5 / gfx1250, wave32)
// ---------------------------------------------------------------------------
typedef __attribute__((ext_vector_type(16))) __bf16        v16bf;
typedef __attribute__((ext_vector_type(8)))  float         v8f;
typedef __attribute__((ext_vector_type(4)))  unsigned int  u32x4;
typedef __attribute__((ext_vector_type(8)))  int           i32x8;
typedef __attribute__((ext_vector_type(4)))  int           i32x4;

struct alignas(16) U4 { unsigned int x, y, z, w; };
struct alignas(8)  U2 { unsigned int x, y; };

union BFrag {           // 16 bf16 = 8 VGPRs per lane (A or B fragment)
    v16bf v;
    U4    q[2];
};

__device__ __forceinline__ unsigned short f2bf(float f) {
    // round-to-nearest-even fp32 -> bf16
    unsigned int u = __float_as_uint(f);
    u += 0x7FFFu + ((u >> 16) & 1u);
    return (unsigned short)(u >> 16);
}

__device__ __forceinline__ unsigned int f2bf2(float a, float b) {
    return (unsigned int)f2bf(a) | ((unsigned int)f2bf(b) << 16);
}

__device__ __forceinline__ v8f wmma_bf16(const BFrag& a, const BFrag& b, v8f c) {
    return __builtin_amdgcn_wmma_f32_16x16x32_bf16(
        /*neg_a=*/false, a.v, /*neg_b=*/false, b.v,
        /*c_mod=*/(short)0, c, /*reuse_a=*/false, /*reuse_b=*/false);
}

// ---------------------------------------------------------------------------
// Tensor Data Mover: async 2D tile load (bf16 elements) global -> LDS.
// Builds a D# per CDNA5 ISA and issues
// tensor_load_to_lds.  Padding codes reproduce the padded LDS row pitch:
//   pad every 2^pad_int * 8 bytes, insert (pad_amt+1) dwords.
// Completion is tracked with TENSORcnt (s_wait_tensorcnt).
// ---------------------------------------------------------------------------
__device__ __forceinline__ void tdm_load_2d_bf16(
    unsigned int lds_off,        // byte offset of tile start in LDS
    const void*  gaddr,          // global address of tile start
    unsigned int tile_w,         // dim0 extent (elements, contiguous)
    unsigned int tile_h,         // dim1 extent (rows)
    unsigned int row_stride,     // tensor_dim0_stride (elements)
    int pad_int, int pad_amt)    // pad codes, -1 = no padding
{
    unsigned long long ga = (unsigned long long)gaddr;
    u32x4 g0;
    g0[0] = 1u;                                              // count=1 (user)
    g0[1] = lds_off;                                         // lds_addr
    g0[2] = (unsigned int)(ga & 0xFFFFFFFFull);              // global_addr lo
    g0[3] = (unsigned int)((ga >> 32) & 0x1FFFFFFull)        // global_addr hi
          | (2u << 30);                                      // type = 2 (image)
    unsigned int dw0 = (1u << 16);                           // data_size=1 (2B)
    if (pad_int >= 0)
        dw0 |= (1u << 20)                                    // pad_enable
             | ((unsigned int)pad_int << 22)                 // pad_interval
             | ((unsigned int)pad_amt << 25);                // pad_amount
    i32x8 g1;
    g1[0] = (int)dw0;
    g1[1] = (int)((tile_w & 0xFFFFu) << 16);                 // tensor_dim0 lo
    g1[2] = (int)(((tile_w >> 16) & 0xFFFFu)                 // tensor_dim0 hi
          | ((tile_h & 0xFFFFu) << 16));                     // tensor_dim1 lo
    g1[3] = (int)(((tile_h >> 16) & 0xFFFFu)                 // tensor_dim1 hi
          | ((tile_w & 0xFFFFu) << 16));                     // tile_dim0
    g1[4] = (int)(tile_h & 0xFFFFu);                         // tile_dim1 (dim2=0)
    g1[5] = (int)row_stride;                                 // dim0_stride lo32
    g1[6] = 0;                                               // stride hi / dim1_stride
    g1[7] = 0;
    i32x4 z4 = {0, 0, 0, 0};
#if defined(__clang_major__) && (__clang_major__ >= 23)
    i32x8 z8 = {0, 0, 0, 0, 0, 0, 0, 0};
    __builtin_amdgcn_tensor_load_to_lds(g0, g1, z4, z4, z8, 0);
#else
    __builtin_amdgcn_tensor_load_to_lds(g0, g1, z4, z4, 0);
#endif
}

__device__ __forceinline__ unsigned int lds_off_of(const void* p) {
    // LDS aperture: generic addr[31:0] is the LDS byte offset
    return (unsigned int)(unsigned long long)p;
}

// ---------------------------------------------------------------------------
// GEMM:  C[M,N] = A[M,K] @ W[N,K]^T + bias[N]
//   A fp32 (converted while staging) or bf16 (TDM async-copied), per template.
//   C bf16 (workspace) or fp32 (d_out), per template.
// 128x128x32 tiles, double-buffered LDS, 8 waves, wave tile 32(M) x 64(N).
// ---------------------------------------------------------------------------
template <int A_IS_F32, int OUT_IS_BF16>
__global__ __launch_bounds__(256) void gemm_bias_kernel(
    const void* __restrict__ Aop,
    const float* __restrict__ W,
    const float* __restrict__ bias,
    void* __restrict__ Cout,
    int M, int N, int K)
{
    constexpr int BM = 128, BN = 128, BK = 32;
    constexpr int LDT = BK + 8;   // padded LDS stride in halves (80B)

    __shared__ __attribute__((aligned(16))) unsigned short sA[2][BM * LDT];
    __shared__ __attribute__((aligned(16))) unsigned short sB[2][BN * LDT];

    const int tid = threadIdx.x;
    const int wid = tid >> 5;          // 0..7
    const int ln  = tid & 31;
    const int g   = ln >> 4;           // lane group (0/1)
    const int lh  = ln & 15;
    const int wm  = wid >> 1;          // 0..3  (M direction, 32 rows each)
    const int wn  = wid & 1;           // 0..1  (N direction, 64 cols each)
    const int m0  = blockIdx.y * BM;
    const int n0  = blockIdx.x * BN;

    v8f acc[2][4];
    const v8f vzero = {0.f,0.f,0.f,0.f,0.f,0.f,0.f,0.f};
#pragma unroll
    for (int i = 0; i < 2; ++i)
#pragma unroll
        for (int j = 0; j < 4; ++j) acc[i][j] = vzero;

    // ---- stage A+W tiles for K-step k0 into buffer `buf` ----
    auto stage_tiles = [&](int buf, int k0) {
        if (A_IS_F32) {
            const float* A = (const float*)Aop;
            const int cl = tid & 7;          // 8 threads/row, 4 floats each
            const int rb = tid >> 3;         // 0..31
#pragma unroll
            for (int i = 0; i < 4; ++i) {
                const int r = rb + 32 * i;
                const float4 f = *(const float4*)&A[(size_t)(m0 + r) * K + k0 + cl * 4];
                U2 p; p.x = f2bf2(f.x, f.y); p.y = f2bf2(f.z, f.w);
                *(U2*)&sA[buf][r * LDT + cl * 4] = p;
            }
        } else {
            // bf16 A: async TDM copy of the 128x32 tile, padded to LDT pitch.
            // pad every 16 dwords (=32 halves) by 4 dwords (=8 halves) -> 40.
            if (wid == 0) {
                const unsigned short* A = (const unsigned short*)Aop;
                tdm_load_2d_bf16(lds_off_of(&sA[buf][0]),
                                 &A[(size_t)m0 * K + k0],
                                 /*tile_w=*/BK, /*tile_h=*/BM,
                                 /*row_stride=*/(unsigned)K,
                                 /*pad_int=*/3, /*pad_amt=*/3);
            }
        }
        {   // W tile: fp32 -> bf16 convert while staging
            const int cl = tid & 7;
            const int rb = tid >> 3;
#pragma unroll
            for (int i = 0; i < 4; ++i) {
                const int r = rb + 32 * i;
                const float4 f = *(const float4*)&W[(size_t)(n0 + r) * K + k0 + cl * 4];
                U2 p; p.x = f2bf2(f.x, f.y); p.y = f2bf2(f.z, f.w);
                *(U2*)&sB[buf][r * LDT + cl * 4] = p;
            }
        }
    };

    // ---- 8 WMMAs from buffer `buf` ----
    auto compute = [&](int buf) {
        BFrag af[2], bfr[4];
#pragma unroll
        for (int mt = 0; mt < 2; ++mt) {
            const int row = wm * 32 + mt * 16 + lh;
            af[mt].q[0] = *(const U4*)&sA[buf][row * LDT + 8 * g];
            af[mt].q[1] = *(const U4*)&sA[buf][row * LDT + 16 + 8 * g];
        }
#pragma unroll
        for (int nt = 0; nt < 4; ++nt) {
            const int row = wn * 64 + nt * 16 + lh;
            bfr[nt].q[0] = *(const U4*)&sB[buf][row * LDT + 8 * g];
            bfr[nt].q[1] = *(const U4*)&sB[buf][row * LDT + 16 + 8 * g];
        }
#pragma unroll
        for (int mt = 0; mt < 2; ++mt)
#pragma unroll
            for (int nt = 0; nt < 4; ++nt)
                acc[mt][nt] = wmma_bf16(af[mt], bfr[nt], acc[mt][nt]);
    };

    stage_tiles(0, 0);
    int cur = 0;
    for (int k0 = 0; k0 < K; k0 += BK) {
        if (!A_IS_F32) {
            if (wid == 0) __builtin_amdgcn_s_wait_tensorcnt(0);
        }
        __syncthreads();                       // staged tiles (ds + TDM) visible
        if (k0 + BK < K) stage_tiles(cur ^ 1, k0 + BK);   // overlap with WMMA
        compute(cur);
        cur ^= 1;
    }

    // ---- epilogue: bias + store ----
#pragma unroll
    for (int nt = 0; nt < 4; ++nt) {
        const int col = n0 + wn * 64 + nt * 16 + lh;
        const float bv = bias[col];
#pragma unroll
        for (int mt = 0; mt < 2; ++mt) {
#pragma unroll
            for (int r = 0; r < 8; ++r) {
                const float v = acc[mt][nt][r] + bv;
                const int row = m0 + wm * 32 + mt * 16 + r + 8 * g;
                if (OUT_IS_BF16)
                    ((unsigned short*)Cout)[(size_t)row * N + col] = f2bf(v);
                else
                    ((float*)Cout)[(size_t)row * N + col] = v;
            }
        }
    }
}

// ---------------------------------------------------------------------------
// Flash attention (causal, GQA 32Q/8KV heads, head_dim=128).
// Grid: (B_*NH_, T_/128). Block 256 = 8 waves; wave handles 16 query rows.
// K tile arrives via TDM async copy; V tile is transposed while staging.
// ---------------------------------------------------------------------------
__global__ __launch_bounds__(256) void attn_kernel(
    const unsigned short* __restrict__ qkv,   // [B,T,QKVD_] bf16
    unsigned short* __restrict__ attn_out)    // [B,T,DIM_]  bf16
{
    constexpr int LDK = HD_ + 8;   // 136 halves  (K tile rows: 32 keys x 128 d)
    constexpr int LDV = 40;        // V^T rows: 128 d x 32 keys
    constexpr int LDP = 40;        // per-wave P tile: 16 x 32

    __shared__ __attribute__((aligned(16))) unsigned short sK[32 * LDK];
    __shared__ __attribute__((aligned(16))) unsigned short sVT[HD_ * LDV];
    __shared__ __attribute__((aligned(16))) unsigned short sP[8][16 * LDP];

    const int bh  = blockIdx.x;
    const int b   = bh >> 5;
    const int h   = bh & 31;
    const int kvh = h >> 2;                 // G = 4
    const int qb  = blockIdx.y;
    const int q0  = qb * 128;

    const int tid = threadIdx.x;
    const int wid = tid >> 5;
    const int ln  = tid & 31;
    const int g   = ln >> 4;
    const int lh  = ln & 15;

    const unsigned short* qp = qkv + (size_t)b * T_ * QKVD_ + (size_t)h * HD_;
    const unsigned short* kp = qkv + (size_t)b * T_ * QKVD_ + DIM_ + (size_t)kvh * HD_;
    const unsigned short* vp = qkv + (size_t)b * T_ * QKVD_ + DIM_ + KVD_ + (size_t)kvh * HD_;

    // ---- Q fragments for this wave's 16 rows (resident all loop long) ----
    BFrag qf[4];
    {
        const int qrow = q0 + wid * 16 + lh;
        const unsigned short* base = qp + (size_t)qrow * QKVD_;
#pragma unroll
        for (int kd = 0; kd < 4; ++kd) {
            qf[kd].q[0] = *(const U4*)(base + kd * 32 + 8 * g);
            qf[kd].q[1] = *(const U4*)(base + kd * 32 + 16 + 8 * g);
        }
    }

    v8f o[8];
    float mrow[8], lrow[8];
    const v8f vzero = {0.f,0.f,0.f,0.f,0.f,0.f,0.f,0.f};
#pragma unroll
    for (int j = 0; j < 8; ++j) o[j] = vzero;
#pragma unroll
    for (int r = 0; r < 8; ++r) { mrow[r] = -INFINITY; lrow[r] = 0.f; }

    const float scale = 0.08838834764831845f;   // 1/sqrt(128)
    const int base_row = q0 + wid * 16 + 8 * g; // first row of this lane's group

    const int nkb = qb * 4 + 4;                 // causal: keys up to q0+127
    for (int kb = 0; kb < nkb; ++kb) {
        const int key0 = kb * 32;
        __syncthreads();                        // prev iter done reading sK/sVT
        // ---- K tile: TDM async copy 32x128 bf16, pad 256B rows to 272B ----
        if (wid == 0) {
            tdm_load_2d_bf16(lds_off_of(&sK[0]),
                             kp + (size_t)key0 * QKVD_,
                             /*tile_w=*/HD_, /*tile_h=*/32,
                             /*row_stride=*/QKVD_,
                             /*pad_int=*/5, /*pad_amt=*/3);
        }
        // ---- V tile: manual transpose into sVT (overlaps the TDM) ----
        {
            const int key = tid >> 3;           // 0..31
            const int cb  = (tid & 7) * 16;     // 0..112
            const unsigned short* vs = vp + (size_t)(key0 + key) * QKVD_ + cb;
            U4 v0v = *(const U4*)(vs);
            U4 v1v = *(const U4*)(vs + 8);
            unsigned short tmp[16];
            *(U4*)&tmp[0] = v0v;
            *(U4*)&tmp[8] = v1v;
#pragma unroll
            for (int i = 0; i < 16; ++i)
                sVT[(cb + i) * LDV + key] = tmp[i];
        }
        if (wid == 0) __builtin_amdgcn_s_wait_tensorcnt(0);
        __syncthreads();

        // ---- S = Q @ K^T  (16 rows x 32 keys, fp32 accum) ----
        v8f s0 = vzero, s1 = vzero;
#pragma unroll
        for (int kd = 0; kd < 4; ++kd) {
            BFrag b0, b1;
            b0.q[0] = *(const U4*)&sK[lh * LDK + kd * 32 + 8 * g];
            b0.q[1] = *(const U4*)&sK[lh * LDK + kd * 32 + 16 + 8 * g];
            b1.q[0] = *(const U4*)&sK[(16 + lh) * LDK + kd * 32 + 8 * g];
            b1.q[1] = *(const U4*)&sK[(16 + lh) * LDK + kd * 32 + 16 + 8 * g];
            s0 = wmma_bf16(qf[kd], b0, s0);
            s1 = wmma_bf16(qf[kd], b1, s1);
        }

        // ---- causal mask, running softmax (fp32) ----
        float p0[8], p1[8];
#pragma unroll
        for (int r = 0; r < 8; ++r) {
            const int row = base_row + r;
            float x0 = s0[r] * scale;
            float x1 = s1[r] * scale;
            if (key0 + lh      > row) x0 = -INFINITY;
            if (key0 + 16 + lh > row) x1 = -INFINITY;
            float mx = fmaxf(x0, x1);
#pragma unroll
            for (int off = 1; off < 16; off <<= 1)
                mx = fmaxf(mx, __shfl_xor(mx, off, 32));
            const float mnew  = fmaxf(mrow[r], mx);
            const float alpha = __expf(mrow[r] - mnew);
            p0[r] = __expf(x0 - mnew);
            p1[r] = __expf(x1 - mnew);
            float rs = p0[r] + p1[r];
#pragma unroll
            for (int off = 1; off < 16; off <<= 1)
                rs += __shfl_xor(rs, off, 32);
            lrow[r] = lrow[r] * alpha + rs;
            mrow[r] = mnew;
#pragma unroll
            for (int j = 0; j < 8; ++j) o[j][r] *= alpha;
        }

        // ---- P: C-layout -> A-fragment layout via per-wave LDS ----
#pragma unroll
        for (int r = 0; r < 8; ++r) {
            sP[wid][(r + 8 * g) * LDP + lh]      = f2bf(p0[r]);
            sP[wid][(r + 8 * g) * LDP + 16 + lh] = f2bf(p1[r]);
        }
        asm volatile("s_wait_dscnt 0" ::: "memory");
        BFrag pf;
        pf.q[0] = *(const U4*)&sP[wid][lh * LDP + 8 * g];
        pf.q[1] = *(const U4*)&sP[wid][lh * LDP + 16 + 8 * g];

        // ---- O += P @ V ----
#pragma unroll
        for (int j = 0; j < 8; ++j) {
            BFrag vf;
            vf.q[0] = *(const U4*)&sVT[(j * 16 + lh) * LDV + 8 * g];
            vf.q[1] = *(const U4*)&sVT[(j * 16 + lh) * LDV + 16 + 8 * g];
            o[j] = wmma_bf16(pf, vf, o[j]);
        }
    }

    // ---- normalize + store (bf16, [B,T,DIM] with head-major cols) ----
    unsigned short* op = attn_out + (size_t)b * T_ * DIM_ + (size_t)h * HD_;
#pragma unroll
    for (int r = 0; r < 8; ++r) {
        const float inv = 1.0f / lrow[r];
        const int row = base_row + r;
#pragma unroll
        for (int j = 0; j < 8; ++j)
            op[(size_t)row * DIM_ + j * 16 + lh] = f2bf(o[j][r] * inv);
    }
}

// ---------------------------------------------------------------------------
// Launch
// ---------------------------------------------------------------------------
extern "C" void kernel_launch(void* const* d_in, const int* in_sizes, int n_in,
                              void* d_out, int out_size, void* d_ws, size_t ws_size,
                              hipStream_t stream) {
    const float* x  = (const float*)d_in[0];   // [B,T,DIM]
    const float* W1 = (const float*)d_in[1];   // [QKVD,DIM]
    const float* b1 = (const float*)d_in[2];   // [QKVD]
    const float* W2 = (const float*)d_in[3];   // [DIM,DIM]
    const float* b2 = (const float*)d_in[4];   // [DIM]
    float* out = (float*)d_out;                // [B,T,DIM] fp32

    unsigned short* qkv  = (unsigned short*)d_ws;              // 25.2 MB bf16
    unsigned short* attn = qkv + (size_t)MTOT_ * QKVD_;        // 16.8 MB bf16

    // 1) qkv = x @ W1^T + b1   (fp32 in, bf16 out)
    gemm_bias_kernel<1, 1><<<dim3(QKVD_ / 128, MTOT_ / 128), 256, 0, stream>>>(
        (const void*)x, W1, b1, (void*)qkv, MTOT_, QKVD_, DIM_);

    // 2) causal GQA flash attention (bf16 in/out, fp32 softmax/accum)
    attn_kernel<<<dim3(B_ * NH_, T_ / 128), 256, 0, stream>>>(qkv, attn);

    // 3) out = attn @ W2^T + b2  (bf16 in, fp32 out)
    gemm_bias_kernel<0, 0><<<dim3(DIM_ / 128, MTOT_ / 128), 256, 0, stream>>>(
        (const void*)attn, W2, b2, (void*)out, MTOT_, DIM_, DIM_);
}
